// RankMixerBlock_83580063580221
// MI455X (gfx1250) — matrix operand order repaired and checked by reading, not compiled, not run on hardware
//
#include <hip/hip_runtime.h>

#define B_   1024
#define T_   8
#define H_   8
#define D_   512
#define E_   4
#define FFN_ 2048

typedef __attribute__((ext_vector_type(8)))  unsigned short u16x8;
typedef __attribute__((ext_vector_type(16))) unsigned short u16x16;
typedef __attribute__((ext_vector_type(16))) __bf16        v16bf;
typedef __attribute__((ext_vector_type(8)))  float         v8f;
typedef __attribute__((ext_vector_type(4)))  float         f32x4;
typedef __attribute__((ext_vector_type(4)))  unsigned int  u32x4;
typedef __attribute__((ext_vector_type(8)))  int           i32x8;
typedef __attribute__((ext_vector_type(4)))  int           i32x4;

union FragU { u16x8 h[2]; u16x16 v; };

__device__ inline unsigned short f2bf(float f) {
  unsigned int u = __float_as_uint(f);
  u += 0x7FFFu + ((u >> 16) & 1u);           // round-to-nearest-even
  return (unsigned short)(u >> 16);
}

__device__ inline float block_reduce_512(float v, float* s) {
  int d = threadIdx.x;
  s[d] = v; __syncthreads();
  #pragma unroll
  for (int off = 256; off >= 1; off >>= 1) {
    if (d < off) s[d] += s[d + off];
    __syncthreads();
  }
  float r = s[0];
  __syncthreads();
  return r;
}

// ---------------------------------------------------------------------------
// TDM: DMA a 128-row x 32-col bf16 tile (row stride = stride_elems) from
// global memory into LDS with 16B padding per 64B row (LDS row = 80 bytes).
// Issued by one wave; completion tracked via TENSORcnt.
// ---------------------------------------------------------------------------
__device__ inline void tdm_load_tile_128x32_bf16(const unsigned short* gsrc,
                                                 unsigned lds_byte_addr,
                                                 unsigned stride_elems) {
  unsigned long long ga = (unsigned long long)(size_t)gsrc;
  u32x4 g0;
  g0[0] = 1u;                                          // count=1 (valid user D#)
  g0[1] = lds_byte_addr;                               // lds_addr
  g0[2] = (unsigned)(ga & 0xFFFFFFFFu);                // global_addr[31:0]
  g0[3] = (unsigned)((ga >> 32) & 0x01FFFFFFu)         // global_addr[56:32]
          | (2u << 30);                                // type=2 ("image")
  i32x8 g1;
  g1[0] = (int)((1u << 16)      // data_size = 2 bytes
              | (1u << 20)      // pad_enable
              | (3u << 22)      // pad_interval: 16 DWORDs (= 64B = 32 bf16)
              | (3u << 25));    // pad_amount:   4 DWORDs (= 16B -> 80B LDS row)
  g1[1] = (int)((32u & 0xFFFFu) << 16);                // tensor_dim0 = 32 (lo16)
  g1[2] = (int)((128u & 0xFFFFu) << 16);               // dim0 hi=0 | tensor_dim1=128 (lo16)
  g1[3] = (int)(32u << 16);                            // dim1 hi=0 | tile_dim0 = 32
  g1[4] = (int)128u;                                   // tile_dim1 = 128, tile_dim2 = 0
  g1[5] = (int)stride_elems;                           // tensor_dim0_stride (lo32)
  g1[6] = 0;                                           // stride hi | dim1_stride lo
  g1[7] = 0;
  i32x4 gz = {0, 0, 0, 0};
#if defined(__clang_major__) && (__clang_major__ >= 23)
  i32x8 gz8 = {0, 0, 0, 0, 0, 0, 0, 0};
  __builtin_amdgcn_tensor_load_to_lds(g0, g1, gz, gz, gz8, 0);
#else
  __builtin_amdgcn_tensor_load_to_lds(g0, g1, gz, gz, 0);
#endif
}

// Stage one 32(K) x 128(N) f32 tile -> bf16, transposed into LDS [n][k].
// Wrow must already point at the N-offset column base.
__device__ inline void stage_B_tile(const float* __restrict__ Wrow,
                                    size_t row_stride, int kk,
                                    unsigned short (*sB)[40], int tid) {
  #pragma unroll
  for (int p = 0; p < 4; p++) {
    int i = tid + p * 256;
    int k = i >> 5, n4 = (i & 31) * 4;
    f32x4 w = *(const f32x4*)&Wrow[(size_t)(kk + k) * row_stride + n4];
    sB[n4 + 0][k] = f2bf(w.x);
    sB[n4 + 1][k] = f2bf(w.y);
    sB[n4 + 2][k] = f2bf(w.z);
    sB[n4 + 3][k] = f2bf(w.w);
  }
}

// 8 WMMAs: this wave's 64x32 sub-tile (4 M-tiles x 2 N-tiles) over one
// K=32 chunk held in LDS.
__device__ inline void wmma_tile(const unsigned short (*sA)[40],
                                 const unsigned short (*sB)[40],
                                 int lane, int wm, int wn, v8f acc[4][2]) {
  int ah = (lane < 16) ? 0 : 8;    // A: lanes 0-15 -> K{0-7,16-23}, 16-31 -> K{8-15,24-31}
  int bh = (lane < 16) ? 0 : 8;    // B: lanes 0-15 -> K{0-7,16-23}, 16-31 -> K{8-15,24-31}
  #pragma unroll
  for (int mi = 0; mi < 4; mi++) {
    int row = wm * 64 + mi * 16 + (lane & 15);
    FragU fa;
    fa.h[0] = *(const u16x8*)&sA[row][ah];
    fa.h[1] = *(const u16x8*)&sA[row][ah + 16];
    v16bf av = __builtin_bit_cast(v16bf, fa.v);
    #pragma unroll
    for (int ni = 0; ni < 2; ni++) {
      int col = wn * 32 + ni * 16 + (lane & 15);
      FragU fb;
      fb.h[0] = *(const u16x8*)&sB[col][bh];
      fb.h[1] = *(const u16x8*)&sB[col][bh + 16];
      v16bf bv = __builtin_bit_cast(v16bf, fb.v);
      acc[mi][ni] = __builtin_amdgcn_wmma_f32_16x16x32_bf16(
          false, av, false, bv, (short)0, acc[mi][ni], false, false);
    }
  }
}

// ---------------------------------------------------------------------------
// Kernel 1: token mixing + residual + LN1 -> x1 (f32 & bf16), router gates,
// aux gate-sum accumulation.  One block per (b,t), 512 threads (= D).
// ---------------------------------------------------------------------------
__global__ __launch_bounds__(512) void k_prep(
    const float* __restrict__ x,
    const float* __restrict__ ln1g, const float* __restrict__ ln1b,
    const float* __restrict__ rw,   const float* __restrict__ rb,
    float* __restrict__ x1f, unsigned short* __restrict__ x1b,
    float* __restrict__ gates, float* __restrict__ auxsum)
{
  __shared__ float sred[512];
  int bt = blockIdx.x;
  int b = bt / T_, t = bt % T_;
  int d = threadIdx.x;

  // mixed[b,t,d] = x[b, d>>6, t*64 + (d&63)]   (H == T, hd = 64)
  float mixed = x[((size_t)(b * T_ + (d >> 6))) * D_ + t * 64 + (d & 63)];
  float v = mixed + x[(size_t)bt * D_ + d];

  float mean = block_reduce_512(v, sred) * (1.0f / D_);
  float c = v - mean;
  float var = block_reduce_512(c * c, sred) * (1.0f / D_);
  float inv = rsqrtf(var + 1e-5f);
  float y = c * inv * ln1g[d] + ln1b[d];

  size_t idx = (size_t)bt * D_ + d;
  x1f[idx] = y;
  x1b[idx] = f2bf(y);

  // router logits (D=512 dot per expert)
  float lg[E_];
  #pragma unroll
  for (int e = 0; e < E_; e++)
    lg[e] = block_reduce_512(y * rw[e * D_ + d], sred) + rb[e];

  if (d == 0) {
    float s = 1e-9f, ge[E_];
    #pragma unroll
    for (int e = 0; e < E_; e++) { ge[e] = fmaxf(lg[e], 0.0f); s += ge[e]; }
    float gs = 0.0f;
    #pragma unroll
    for (int e = 0; e < E_; e++) {
      float g = ge[e] / s;
      gates[bt * E_ + e] = g;
      gs += g;
    }
    atomicAdd(auxsum, gs);
  }
}

// ---------------------------------------------------------------------------
// Kernel 2: hidden = gelu(x1 @ W1[t,e] + eb1) * gate, stored bf16 as
// Hs[t][b][e*FFN+f].  M tile 128, N tile 128, K chunk 32, double-buffered:
// TDM DMA of A chunk k+1 + float4 B staging overlap the 8 WMMAs of chunk k.
// ---------------------------------------------------------------------------
__global__ __launch_bounds__(256) void k_gemm1(
    const unsigned short* __restrict__ x1b,
    const float* __restrict__ ew1, const float* __restrict__ eb1,
    const float* __restrict__ gates, unsigned short* __restrict__ hs)
{
  __shared__ __align__(16) unsigned short sA[2][128][40];  // ping-pong, 80B rows
  __shared__ __align__(16) unsigned short sB[2][128][40];  // transposed [n][k]

  int nt = blockIdx.x;            // 16 tiles of FFN
  int mt = blockIdx.y;            // 8 tiles of B
  int te = blockIdx.z;            // t*4 + e
  int t = te >> 2, e = te & 3;
  const float* W1col = ew1 + ((size_t)(t * E_ + e)) * D_ * FFN_ + nt * 128;
  const unsigned short* Abase =
      x1b + ((size_t)(mt * 128) * T_ + t) * D_;            // row stride T_*D_

  int tid = threadIdx.x;
  int lane = tid & 31, wave = tid >> 5;
  int wm = wave & 1, wn = wave >> 1;
  unsigned sA_lds[2] = { (unsigned)(size_t)&sA[0][0][0],
                         (unsigned)(size_t)&sA[1][0][0] };

  v8f acc[4][2] = {};

  // prologue: fill buffer 0
  if (wave == 0) tdm_load_tile_128x32_bf16(Abase, sA_lds[0], T_ * D_);
  stage_B_tile(W1col, FFN_, 0, sB[0], tid);
  if (wave == 0) __builtin_amdgcn_s_wait_tensorcnt(0);
  __syncthreads();

  int buf = 0;
  for (int kk = 0; kk < D_; kk += 32) {
    int nxt = buf ^ 1;
    if (kk + 32 < D_) {                     // prefetch next chunk
      if (wave == 0)
        tdm_load_tile_128x32_bf16(Abase + kk + 32, sA_lds[nxt], T_ * D_);
      stage_B_tile(W1col, FFN_, kk + 32, sB[nxt], tid);
    }
    wmma_tile(sA[buf], sB[buf], lane, wm, wn, acc);
    if (wave == 0) __builtin_amdgcn_s_wait_tensorcnt(0);
    __syncthreads();
    buf = nxt;
  }

  // epilogue: + eb1, GELU(exact), * gate, -> bf16
  int rowoff = (lane < 16) ? 0 : 8;
  #pragma unroll
  for (int mi = 0; mi < 4; mi++) {
    float g[8];
    #pragma unroll
    for (int r = 0; r < 8; r++) {
      int m = mt * 128 + wm * 64 + mi * 16 + rowoff + r;
      g[r] = gates[((size_t)m * T_ + t) * E_ + e];
    }
    #pragma unroll
    for (int ni = 0; ni < 2; ni++) {
      int n = nt * 128 + wn * 32 + ni * 16 + (lane & 15);
      float bias = eb1[((size_t)(t * E_ + e)) * FFN_ + n];
      #pragma unroll
      for (int r = 0; r < 8; r++) {
        int m = mt * 128 + wm * 64 + mi * 16 + rowoff + r;
        float h = acc[mi][ni][r] + bias;
        h = 0.5f * h * (1.0f + erff(h * 0.70710678f));
        hs[((size_t)t * B_ + m) * (E_ * FFN_) + e * FFN_ + n] = f2bf(h * g[r]);
      }
    }
  }
}

// ---------------------------------------------------------------------------
// Kernel 3: ffn_out[b,t,:] = Hs[t][b] @ W2flat[t]  (K = E*FFN = 8192, gates
// already folded in).  Same double-buffered pipeline.
// ---------------------------------------------------------------------------
__global__ __launch_bounds__(256) void k_gemm2(
    const unsigned short* __restrict__ hs,
    const float* __restrict__ ew2, float* __restrict__ ffn)
{
  __shared__ __align__(16) unsigned short sA[2][128][40];
  __shared__ __align__(16) unsigned short sB[2][128][40];

  int nt = blockIdx.x;            // 4 tiles of D
  int mt = blockIdx.y;            // 8 tiles of B
  int t  = blockIdx.z;
  const float* W2col = ew2 + (size_t)t * E_ * FFN_ * D_ + nt * 128;  // [8192][512]
  const unsigned short* Abase =
      hs + (size_t)t * B_ * E_ * FFN_ + (size_t)(mt * 128) * (E_ * FFN_);

  int tid = threadIdx.x;
  int lane = tid & 31, wave = tid >> 5;
  int wm = wave & 1, wn = wave >> 1;
  unsigned sA_lds[2] = { (unsigned)(size_t)&sA[0][0][0],
                         (unsigned)(size_t)&sA[1][0][0] };

  v8f acc[4][2] = {};

  if (wave == 0) tdm_load_tile_128x32_bf16(Abase, sA_lds[0], E_ * FFN_);
  stage_B_tile(W2col, D_, 0, sB[0], tid);
  if (wave == 0) __builtin_amdgcn_s_wait_tensorcnt(0);
  __syncthreads();

  int buf = 0;
  for (int kk = 0; kk < E_ * FFN_; kk += 32) {
    int nxt = buf ^ 1;
    if (kk + 32 < E_ * FFN_) {
      if (wave == 0)
        tdm_load_tile_128x32_bf16(Abase + kk + 32, sA_lds[nxt], E_ * FFN_);
      stage_B_tile(W2col, D_, kk + 32, sB[nxt], tid);
    }
    wmma_tile(sA[buf], sB[buf], lane, wm, wn, acc);
    if (wave == 0) __builtin_amdgcn_s_wait_tensorcnt(0);
    __syncthreads();
    buf = nxt;
  }

  int rowoff = (lane < 16) ? 0 : 8;
  #pragma unroll
  for (int mi = 0; mi < 4; mi++) {
    #pragma unroll
    for (int ni = 0; ni < 2; ni++) {
      int n = nt * 128 + wn * 32 + ni * 16 + (lane & 15);
      #pragma unroll
      for (int r = 0; r < 8; r++) {
        int m = mt * 128 + wm * 64 + mi * 16 + rowoff + r;
        ffn[((size_t)m * T_ + t) * D_ + n] = acc[mi][ni][r];
      }
    }
  }
}

// ---------------------------------------------------------------------------
// Kernel 4: out = LN2(ffn_out + sum_e gate*eb2 + x1)
// ---------------------------------------------------------------------------
__global__ __launch_bounds__(512) void k_final(
    const float* __restrict__ ffn, const float* __restrict__ x1f,
    const float* __restrict__ gates, const float* __restrict__ eb2,
    const float* __restrict__ ln2g, const float* __restrict__ ln2b,
    float* __restrict__ out)
{
  __shared__ float sred[512];
  int bt = blockIdx.x;
  int t = bt % T_;
  int d = threadIdx.x;
  size_t idx = (size_t)bt * D_ + d;

  float v = ffn[idx] + x1f[idx];
  #pragma unroll
  for (int e = 0; e < E_; e++)
    v += gates[bt * E_ + e] * eb2[((size_t)(t * E_ + e)) * D_ + d];

  float mean = block_reduce_512(v, sred) * (1.0f / D_);
  float c = v - mean;
  float var = block_reduce_512(c * c, sred) * (1.0f / D_);
  float inv = rsqrtf(var + 1e-5f);
  out[idx] = c * inv * ln2g[d] + ln2b[d];
}

__global__ void k_aux(const float* __restrict__ auxsum, float* __restrict__ out) {
  out[(size_t)B_ * T_ * D_] = 0.01f * auxsum[0] / (float)(B_ * T_);
}

// ---------------------------------------------------------------------------
extern "C" void kernel_launch(void* const* d_in, const int* in_sizes, int n_in,
                              void* d_out, int out_size, void* d_ws, size_t ws_size,
                              hipStream_t stream) {
  const float* x       = (const float*)d_in[0];
  const float* ln1g    = (const float*)d_in[1];
  const float* ln1b    = (const float*)d_in[2];
  const float* rw      = (const float*)d_in[3];
  const float* rb      = (const float*)d_in[4];
  const float* ew1     = (const float*)d_in[5];
  const float* eb1     = (const float*)d_in[6];
  const float* ew2     = (const float*)d_in[7];
  const float* eb2     = (const float*)d_in[8];
  const float* ln2g    = (const float*)d_in[9];
  const float* ln2b    = (const float*)d_in[10];
  float* out = (float*)d_out;

  // workspace carve-up (256B aligned)
  char* ws = (char*)d_ws;
  size_t o = 0;
  auto take = [&](size_t bytes) { size_t r = o; o = (o + bytes + 255) & ~(size_t)255; return r; };
  float*          x1f  = (float*)         (ws + take((size_t)B_ * T_ * D_ * 4));
  unsigned short* x1b  = (unsigned short*)(ws + take((size_t)B_ * T_ * D_ * 2));
  float*          gts  = (float*)         (ws + take((size_t)B_ * T_ * E_ * 4));
  float*          aux  = (float*)         (ws + take(4));
  unsigned short* hs   = (unsigned short*)(ws + take((size_t)T_ * B_ * E_ * FFN_ * 2));
  float*          ffn  = (float*)         (ws + take((size_t)B_ * T_ * D_ * 4));
  (void)ws_size; (void)in_sizes; (void)n_in; (void)out_size;

  hipMemsetAsync(aux, 0, sizeof(float), stream);

  k_prep<<<dim3(B_ * T_), dim3(512), 0, stream>>>(x, ln1g, ln1b, rw, rb,
                                                  x1f, x1b, gts, aux);
  k_gemm1<<<dim3(FFN_ / 128, B_ / 128, T_ * E_), dim3(256), 0, stream>>>(
      x1b, ew1, eb1, gts, hs);
  k_gemm2<<<dim3(D_ / 128, B_ / 128, T_), dim3(256), 0, stream>>>(hs, ew2, ffn);
  k_final<<<dim3(B_ * T_), dim3(512), 0, stream>>>(ffn, x1f, gts, eb2,
                                                   ln2g, ln2b, out);
  k_aux<<<dim3(1), dim3(1), 0, stream>>>(aux, out);
}
